// GraphCNN_11175504904202
// MI455X (gfx1250) — matrix operand order repaired and checked by reading, not compiled, not run on hardware
//
#include <hip/hip_runtime.h>
#include <hip/hip_bf16.h>

typedef __attribute__((ext_vector_type(16))) _Float16 v16h;
typedef __attribute__((ext_vector_type(4)))  _Float16 v4h;
typedef __attribute__((ext_vector_type(8)))  float    v8f;

#define Bg 8
#define Nn0 2048
#define Nn1 1024
#define Nn2 512
#define Dd 128
#define E0 32768
#define E1 16384
#define E2 8192
#define OUTD 10

#define TM 128   // block tile rows
#define KS 32    // K step per WMMA pass
#define NN 128   // fixed N (feature dim)

// ---------------------------------------------------------------------------
// C[M x 128] = A[M x K] @ B[K x 128] via f16 WMMA, f32 accumulate.
// Batched over gridDim.y with per-operand strides. BIAS/RELU/AUX compiled in.
//
// LDS holds tiles pre-swizzled into the exact WMMA fragment layout
// (ISA §7.12.2), so each wave fetches fragments as contiguous 32B reads
// (ds_load_b128 pairs) with zero repacking VALU.
//
//   A frag (16x32 f16): lane = (row&15) + 16*hi,  elem e:
//       K = hi*8+e (e<8) | 16+hi*8+(e-8) (e>=8)
//       inverse: hi=(K>>3)&1, e=(K&7)+((K>>4)<<3)
//   B frag (32x16 f16): lane = (col&15) + 16*((K>>3)&1), elem e = (K&7)+((K>>4)<<3) (same K map as A)
// ---------------------------------------------------------------------------
template<bool BIAS, bool RELU, bool AUX>
__launch_bounds__(256)
__global__ void gemm_wmma_f16(const float* __restrict__ A, int lda, long strideA,
                              const float* __restrict__ Bm, long strideB,
                              float* __restrict__ C, long strideC,
                              const float* __restrict__ bias,
                              float* __restrict__ aux, long strideAux,
                              int K)
{
    __shared__ __align__(32) _Float16 Afrag[8][32][16];  // [wave][lane][elem] 8KB
    __shared__ __align__(32) _Float16 Bfrag[8][32][16];  // [ntile][lane][elem] 8KB

    const int b = blockIdx.y;
    const int m0blk = blockIdx.x * TM;
    const float* Ab = A + (long)b * strideA;
    const float* Bb = Bm + (long)b * strideB;

    const int tid  = threadIdx.x;
    const int wave = tid >> 5;
    const int lane = tid & 31;
    const int half = lane & 15;
    const int hiL  = lane >> 4;

    v8f acc[8];
#pragma unroll
    for (int i = 0; i < 8; ++i) acc[i] = (v8f){0,0,0,0,0,0,0,0};

    for (int k0 = 0; k0 < K; k0 += KS) {
        if (k0 + KS < K) {
            // prefetch next tiles (global_prefetch_b8)
            __builtin_prefetch(Ab + (long)(m0blk + (tid >> 1)) * lda + k0 + KS + (tid & 1) * 16, 0, 1);
            __builtin_prefetch(Bb + (long)(k0 + KS + (tid >> 3)) * NN + ((tid & 7) << 4), 0, 1);
        }
        // --- stage A tile: 128 rows x 32 K, f32 -> f16, fragment layout ---
#pragma unroll
        for (int i = 0; i < 4; ++i) {
            int idx = tid + i * 256;                    // 1024 quad slots
            int row = idx >> 3;
            int k   = (idx & 7) * 4;
            const float4 v = *(const float4*)(Ab + (long)(m0blk + row) * lda + k0 + k);
            int hi = (k >> 3) & 1;
            int eb = (k & 7) + ((k >> 4) << 3);
            *(v4h*)&Afrag[row >> 4][(row & 15) + (hi << 4)][eb] =
                (v4h){(_Float16)v.x, (_Float16)v.y, (_Float16)v.z, (_Float16)v.w};
        }
        // --- stage B tile: 32 K x 128 cols, column-quad per thread ---
#pragma unroll
        for (int i = 0; i < 4; ++i) {
            int idx = tid + i * 256;                    // 1024 quad slots
            int col = idx & 127;
            int k   = (idx >> 7) * 4;
            const float* bp = Bb + (long)(k0 + k) * NN + col;
            float b0 = bp[0 * NN], b1 = bp[1 * NN], b2 = bp[2 * NN], b3 = bp[3 * NN];
            *(v4h*)&Bfrag[col >> 4][(col & 15) + (((k >> 3) & 1) << 4)][(k & 7) + ((k >> 4) << 3)] =
                (v4h){(_Float16)b0, (_Float16)b1, (_Float16)b2, (_Float16)b3};
        }
        __syncthreads();

        const v16h af = *(const v16h*)&Afrag[wave][lane][0];
#pragma unroll
        for (int nt = 0; nt < 8; ++nt) {
            const v16h bf = *(const v16h*)&Bfrag[nt][lane][0];
            acc[nt] = __builtin_amdgcn_wmma_f32_16x16x32_f16(
                false, af, false, bf, (short)0, acc[nt], false, false);
        }
        __syncthreads();
    }

    // epilogue: C/D layout — lane<16: col=lane, M=r; lane>=16: M=r+8
    const int rowbase = m0blk + wave * 16 + hiL * 8;
    float* cbase = C + (long)b * strideC + (long)rowbase * NN + half;
    float* abase = AUX ? (aux + (long)b * strideAux + (long)rowbase * NN + half) : nullptr;

    float bv[8];
    if (BIAS) {
#pragma unroll
        for (int nt = 0; nt < 8; ++nt) bv[nt] = bias[nt * 16 + half];
    }
#pragma unroll
    for (int nt = 0; nt < 8; ++nt) {
#pragma unroll
        for (int r = 0; r < 8; ++r) {
            float v = acc[nt][r];
            if (BIAS) v += bv[nt];
            if (RELU) v = fmaxf(v, 0.0f);
            cbase[r * NN + nt * 16] = v;
            if (AUX) abase[r * NN + nt * 16] = v;
        }
    }
}

// ---------------------------------------------------------------------------
// Sparse / pointwise helper kernels
// ---------------------------------------------------------------------------
__global__ void zero_f32(float* p, int n) {
    int i = blockIdx.x * blockDim.x + threadIdx.x;
    if (i < n) p[i] = 0.0f;
}

__global__ void count_deg(const int* __restrict__ ei, float* __restrict__ deg,
                          int E, int n, int total) {
    int i = blockIdx.x * blockDim.x + threadIdx.x;
    if (i >= total) return;
    int b = i / E, e = i - b * E;
    int dst = ei[(long)b * 2 * E + E + e];
    atomicAdd(deg + b * n + dst, 1.0f);
}

__global__ void rsqrt_k(float* p, int n) {
    int i = blockIdx.x * blockDim.x + threadIdx.x;
    if (i < n) p[i] = rsqrtf(p[i] + 1.0f);   // +1 self loop
}

// h = xw * dinv^2 + bias (GCN self-loop term; also seeds the output buffer)
__global__ void gcn_init(const float* __restrict__ xw, const float* __restrict__ dinv,
                         const float* __restrict__ bias, float* __restrict__ h, int total) {
    int i = blockIdx.x * blockDim.x + threadIdx.x;
    if (i >= total) return;
    int d = i & 127;
    int node = i >> 7;                 // flat (b*N + n)
    float di = dinv[node];
    h[i] = xw[i] * di * di + bias[d];
}

__global__ void gcn_scatter(const int* __restrict__ ei, const float* __restrict__ xw,
                            const float* __restrict__ dinv, float* __restrict__ h,
                            int E, int n, int total) {
    int i = blockIdx.x * blockDim.x + threadIdx.x;
    if (i >= total) return;
    int d = i & 127;
    int rest = i >> 7;                 // b*E + e
    int b = rest / E, e = rest - b * E;
    const int* eb = ei + (long)b * 2 * E;
    int src = eb[e], dst = eb[E + e];
    float norm = dinv[b * n + src] * dinv[b * n + dst];
    atomicAdd(h + ((long)(b * n + dst) << 7) + d,
              xw[((long)(b * n + src) << 7) + d] * norm);
}

__global__ void gin_scatter(const int* __restrict__ ei, const float* __restrict__ p,
                            float* __restrict__ agg, int E, int n, int total) {
    int i = blockIdx.x * blockDim.x + threadIdx.x;
    if (i >= total) return;
    int d = i & 127;
    int rest = i >> 7;
    int b = rest / E, e = rest - b * E;
    const int* eb = ei + (long)b * 2 * E;
    int src = eb[e], dst = eb[E + e];
    atomicAdd(agg + ((long)(b * n + dst) << 7) + d,
              p[((long)(b * n + src) << 7) + d]);
}

// rep[b] = h2^T @ (h2 @ w_att), h2: [512 x 128] per graph
__global__ void att_pool(const float* __restrict__ h2, const float* __restrict__ w_att,
                         float* __restrict__ rep) {
    __shared__ float s[Nn2];
    int b = blockIdx.x;
    const float* H = h2 + (long)b * Nn2 * Dd;
    for (int i = threadIdx.x; i < Nn2; i += blockDim.x) {
        float acc = 0.0f;
        for (int d = 0; d < Dd; ++d) acc += H[i * Dd + d] * w_att[d];
        s[i] = acc;
    }
    __syncthreads();
    for (int d = threadIdx.x; d < Dd; d += blockDim.x) {
        float acc = 0.0f;
        for (int i = 0; i < Nn2; ++i) acc += H[i * Dd + d] * s[i];
        rep[b * Dd + d] = acc;
    }
}

__global__ void final_lin(const float* __restrict__ rep, const float* __restrict__ w,
                          const float* __restrict__ bias, float* __restrict__ out) {
    int i = threadIdx.x;
    if (i >= Bg * OUTD) return;
    int b = i / OUTD, o = i - b * OUTD;
    float acc = bias[o];
    for (int d = 0; d < Dd; ++d) acc += rep[b * Dd + d] * w[d * OUTD + o];
    out[i] = acc;
}

// ---------------------------------------------------------------------------
extern "C" void kernel_launch(void* const* d_in, const int* in_sizes, int n_in,
                              void* d_out, int out_size, void* d_ws, size_t ws_size,
                              hipStream_t stream) {
    const float* x     = (const float*)d_in[0];
    const float* rt1   = (const float*)d_in[1];
    const float* rt2   = (const float*)d_in[2];
    const float* gcn_w = (const float*)d_in[3];
    const float* gcn_b = (const float*)d_in[4];
    const float* g1w1  = (const float*)d_in[5];
    const float* g1b1  = (const float*)d_in[6];
    const float* g1w2  = (const float*)d_in[7];
    const float* g1b2  = (const float*)d_in[8];
    const float* g2w1  = (const float*)d_in[9];
    const float* g2b1  = (const float*)d_in[10];
    const float* g2w2  = (const float*)d_in[11];
    const float* g2b2  = (const float*)d_in[12];
    const float* w_att = (const float*)d_in[13];
    const float* lin_w = (const float*)d_in[14];
    const float* lin_b = (const float*)d_in[15];
    const int*   ei0   = (const int*)d_in[16];
    const int*   ei1   = (const int*)d_in[17];
    const int*   ei2   = (const int*)d_in[18];
    float* out = (float*)d_out;

    // workspace carve-up (floats)
    float* ws   = (float*)d_ws;
    float* dinv = ws;  ws += Bg * Nn0;
    float* xw   = ws;  ws += Bg * Nn0 * Dd;
    float* hg   = ws;  ws += Bg * Nn0 * Dd;
    float* p1   = ws;  ws += Bg * Nn1 * Dd;
    float* agg1 = ws;  ws += Bg * Nn1 * Dd;
    float* t1   = ws;  ws += Bg * Nn1 * Dd;
    float* h1   = ws;  ws += Bg * Nn1 * Dd;
    float* p2   = ws;  ws += Bg * Nn2 * Dd;
    float* agg2 = ws;  ws += Bg * Nn2 * Dd;
    float* t2   = ws;  ws += Bg * Nn2 * Dd;
    float* h2   = ws;  ws += Bg * Nn2 * Dd;
    float* rep  = ws;  ws += Bg * Dd;

    const int T = 256;
    // 1) degree -> dinv
    zero_f32<<<(Bg*Nn0 + T-1)/T, T, 0, stream>>>(dinv, Bg*Nn0);
    count_deg<<<(Bg*E0 + T-1)/T, T, 0, stream>>>(ei0, dinv, E0, Nn0, Bg*E0);
    rsqrt_k<<<(Bg*Nn0 + T-1)/T, T, 0, stream>>>(dinv, Bg*Nn0);

    // 2) xw = x @ gcn_w  (M=2048, K=128)
    gemm_wmma_f16<false,false,false><<<dim3(Nn0/TM, Bg), T, 0, stream>>>(
        x, Dd, (long)Nn0*Dd, gcn_w, 0, xw, (long)Nn0*Dd, nullptr, nullptr, 0, Dd);

    // 3) GCN: self-loop init then normalized edge scatter -> hg
    gcn_init<<<(Bg*Nn0*Dd + T-1)/T, T, 0, stream>>>(xw, dinv, gcn_b, hg, Bg*Nn0*Dd);
    gcn_scatter<<<(Bg*E0*Dd + T-1)/T, T, 0, stream>>>(ei0, xw, dinv, hg, E0, Nn0, Bg*E0*Dd);

    // 4) p1 = rt1 @ hg  (M=1024, K=2048), aux seeds agg1 = p1
    gemm_wmma_f16<false,false,true><<<dim3(Nn1/TM, Bg), T, 0, stream>>>(
        rt1, Nn0, (long)Nn1*Nn0, hg, (long)Nn0*Dd, p1, (long)Nn1*Dd,
        nullptr, agg1, (long)Nn1*Dd, Nn0);

    // 5) GIN1: agg1 += scatter(p1); t1 = relu(agg1@w1+b1); h1 = relu(t1@w2+b2)
    gin_scatter<<<(Bg*E1*Dd + T-1)/T, T, 0, stream>>>(ei1, p1, agg1, E1, Nn1, Bg*E1*Dd);
    gemm_wmma_f16<true,true,false><<<dim3(Nn1/TM, Bg), T, 0, stream>>>(
        agg1, Dd, (long)Nn1*Dd, g1w1, 0, t1, (long)Nn1*Dd, g1b1, nullptr, 0, Dd);
    gemm_wmma_f16<true,true,false><<<dim3(Nn1/TM, Bg), T, 0, stream>>>(
        t1, Dd, (long)Nn1*Dd, g1w2, 0, h1, (long)Nn1*Dd, g1b2, nullptr, 0, Dd);

    // 6) p2 = rt2 @ h1  (M=512, K=1024), aux seeds agg2
    gemm_wmma_f16<false,false,true><<<dim3(Nn2/TM, Bg), T, 0, stream>>>(
        rt2, Nn1, (long)Nn2*Nn1, h1, (long)Nn1*Dd, p2, (long)Nn2*Dd,
        nullptr, agg2, (long)Nn2*Dd, Nn1);

    // 7) GIN2
    gin_scatter<<<(Bg*E2*Dd + T-1)/T, T, 0, stream>>>(ei2, p2, agg2, E2, Nn2, Bg*E2*Dd);
    gemm_wmma_f16<true,true,false><<<dim3(Nn2/TM, Bg), T, 0, stream>>>(
        agg2, Dd, (long)Nn2*Dd, g2w1, 0, t2, (long)Nn2*Dd, g2b1, nullptr, 0, Dd);
    gemm_wmma_f16<true,true,false><<<dim3(Nn2/TM, Bg), T, 0, stream>>>(
        t2, Dd, (long)Nn2*Dd, g2w2, 0, h2, (long)Nn2*Dd, g2b2, nullptr, 0, Dd);

    // 8) attention pooling + final linear
    att_pool<<<Bg, 256, 0, stream>>>(h2, w_att, rep);
    final_lin<<<1, 128, 0, stream>>>(rep, lin_w, lin_b, out);
}
